// PointNet_5892695130701
// MI455X (gfx1250) — matrix operand order repaired and checked by reading, not compiled, not run on hardware
//
#include <hip/hip_runtime.h>
#include <hip/hip_bf16.h>

// ---------------------------------------------------------------------------
// PointNet forward on gfx1250 (MI455X).
// All dense layers -> v_wmma_f32_16x16x32_bf16 (f32 accumulate).
// Activations flow as PACKED bf16 pairs (uint = 2 x bf16 along K), weights are
// prepacked once per launch. GEMM blocks are 32 rows x (64*NT) cols with one
// A-fragment reused across NT WMMAs per wave, LDS double-buffered (one barrier
// per K-step). The N x 1024 layers are fused with segment_max (f32-bits uint
// atomicMax; valid since post-ReLU >= 0) so the 256MB intermediates never
// touch HBM, and wide tiles (NT=4) cut X re-reads from 16x to 4x.
// ---------------------------------------------------------------------------

typedef __attribute__((ext_vector_type(16))) __bf16 v16bf;
typedef __attribute__((ext_vector_type(8)))  float  v8f;

union FragBF { v16bf v; unsigned u[8]; };

__device__ __forceinline__ unsigned bfpack(float a, float b) {
    // round-to-nearest-even f32 -> bf16, packed low|high
    unsigned ua = __float_as_uint(a);
    unsigned ub = __float_as_uint(b);
    ua = (ua + 0x7FFFu + ((ua >> 16) & 1u)) >> 16;
    ub = (ub + 0x7FFFu + ((ub >> 16) & 1u)) >> 16;
    return (ua & 0xFFFFu) | (ub << 16);
}

// ---------------------------------------------------------------------------
// Fused GEMM on packed-bf16 operands.
//   X packed [M, K/2] uints, W prepacked [K/2, N] uints (pairs along K).
//   NT    : column tiles per wave; block tile = 32 x (64*NT)
//   BN_   : eval-mode BatchNorm1d     RELU_: ReLU
//   SEG_  : no Y store; per-graph (4096 rows) max -> gmax via uint atomicMax
//   PG_   : per-graph weights: W base += graph * (K/2)*N  (feature transform)
//   MG_   : M may be < 32 (row-clamped loads, guarded stores)
//   PK_   : store Y as packed bf16 pairs (else f32)
// Block = 256 threads = 8 waves. K, N multiples of 32 / 64*NT by construction.
// ---------------------------------------------------------------------------
template <int NT, bool BN_, bool RELU_, bool SEG_, bool PG_, bool MG_, bool PK_>
__global__ __launch_bounds__(256) void gemm_wmma(
    const unsigned* __restrict__ Xp, const unsigned* __restrict__ Wp,
    const float* __restrict__ bias,
    const float* __restrict__ gamma, const float* __restrict__ beta,
    const float* __restrict__ mean,  const float* __restrict__ var,
    unsigned* __restrict__ Yp, float* __restrict__ Yf,
    unsigned* __restrict__ gmax,
    int M, int K, int N)
{
    __shared__ unsigned sX[2][32][16];        // 32 rows x 16 K-pairs
    __shared__ unsigned sW[2][16][64 * NT];   // 16 K-pairs x 64*NT cols

    const int tid   = threadIdx.x;
    const int lane  = tid & 31;
    const int wave  = tid >> 5;
    const int mtile = wave >> 2;              // 0..1
    const int ng    = wave & 3;               // wave column group
    const int hi    = lane >> 4;
    const int lm    = lane & 15;

    const int rowBase = blockIdx.x * 32;
    const int colBase = blockIdx.y * 64 * NT;
    const int graph   = rowBase >> 12;        // 4096 points / graph
    const int K2      = K >> 1;
    const unsigned* Wb = Wp + (PG_ ? (size_t)graph * K2 * N : 0);

    // X-stage coords: one uint2 per thread (32x16 slab)
    const int xr  = tid >> 3;
    const int xkp = (tid & 7) << 1;
    int xrow = rowBase + xr;
    if (MG_) xrow = min(xrow, M - 1);
    const unsigned* xsrc = Xp + (size_t)xrow * K2 + xkp;

    // W-stage coords: NT uint4s per thread (16 x 64*NT slab)
    const int wkp = tid >> 4;
    const int wc0 = (tid & 15) << 2;
    const unsigned* wsrc = Wb + (size_t)wkp * N + colBase + wc0;

    auto stage = [&](int k0, int buf) {
        uint2 xv = *(const uint2*)(xsrc + (k0 >> 1));
        sX[buf][xr][xkp]     = xv.x;
        sX[buf][xr][xkp + 1] = xv.y;
        const unsigned* wrow = wsrc + (size_t)(k0 >> 1) * N;
        #pragma unroll
        for (int j = 0; j < NT; ++j) {
            uint4 wv = *(const uint4*)(wrow + j * 64);
            sW[buf][wkp][wc0 + j * 64 + 0] = wv.x;
            sW[buf][wkp][wc0 + j * 64 + 1] = wv.y;
            sW[buf][wkp][wc0 + j * 64 + 2] = wv.z;
            sW[buf][wkp][wc0 + j * 64 + 3] = wv.w;
        }
    };

    v8f acc[NT] = {};

    stage(0, 0);
    __syncthreads();
    int buf = 0;
    for (int k0 = 0; k0 < K; k0 += 32) {
        if (k0 + 32 < K) stage(k0 + 32, buf ^ 1);   // prefetch next slab

        FragBF a;
        #pragma unroll
        for (int r = 0; r < 8; ++r) {
            int pa = (r < 4 ? r : r + 4) + (hi ? 4 : 0);   // A K-pair index
            a.u[r] = sX[buf][mtile * 16 + lm][pa];
        }
        #pragma unroll
        for (int nt = 0; nt < NT; ++nt) {
            FragBF b;
            #pragma unroll
            for (int r = 0; r < 8; ++r) {
                int pb = (r < 4 ? r : r + 4) + (hi ? 4 : 0);   // B K-pair index
                b.u[r] = sW[buf][pb][ng * 16 * NT + nt * 16 + lm];
            }
            acc[nt] = __builtin_amdgcn_wmma_f32_16x16x32_bf16(
                false, a.v, false, b.v, (short)0, acc[nt], false, false);
        }
        __syncthreads();
        buf ^= 1;
    }

    // ---- epilogue ----
    #pragma unroll
    for (int nt = 0; nt < NT; ++nt) {
        const int col = colBase + ng * 16 * NT + nt * 16 + lm;
        float add = 0.f, scale = 1.f, shift = 0.f;
        if (!PG_) add = bias[col];
        if (BN_) {
            scale = gamma[col] * rsqrtf(var[col] + 1e-5f);
            shift = beta[col] - mean[col] * scale;
        }
        float vmax = 0.f;
        #pragma unroll
        for (int r = 0; r < 8; ++r) {
            int row = rowBase + mtile * 16 + hi * 8 + r;
            float v = acc[nt][r] + add;
            if (BN_)   v = v * scale + shift;
            if (RELU_) v = fmaxf(v, 0.f);
            if (SEG_) {
                vmax = fmaxf(vmax, v);
            } else if (PK_) {
                float pv = __shfl_xor(v, 1, 32);           // partner column
                if (!(lane & 1) && (!MG_ || row < M))
                    Yp[(size_t)row * (N >> 1) + (col >> 1)] = bfpack(v, pv);
            } else {
                if (!MG_ || row < M)
                    Yf[(size_t)row * N + col] = v;
            }
        }
        if (SEG_) {
            vmax = fmaxf(vmax, __shfl_xor(vmax, 16, 32));  // lanes l <-> l+16
            if (hi == 0)
                atomicMax(&gmax[graph * N + col], __float_as_uint(vmax));
        }
    }
}

// ---------------------------------------------------------------------------
// Pack / prepack helpers
// ---------------------------------------------------------------------------
__global__ void fill_zero(float* p, int n) {
    int i = blockIdx.x * blockDim.x + threadIdx.x;
    if (i < n) p[i] = 0.f;
}

// f32 [M,K] -> packed [M,K/2]
__global__ void pack_rows(const float* __restrict__ x, unsigned* __restrict__ xp, int total2) {
    int i = blockIdx.x * blockDim.x + threadIdx.x;
    if (i < total2) xp[i] = bfpack(x[2 * i], x[2 * i + 1]);
}

// W[K,N] f32 -> packed [Kpad/2, N] (zero-padded in K)
__global__ void prepack_w(const float* __restrict__ w, unsigned* __restrict__ wp,
                          int Kreal, int Kpad, int N) {
    int i = blockIdx.x * blockDim.x + threadIdx.x;
    int total = (Kpad >> 1) * N;
    if (i >= total) return;
    int kp = i / N, n = i - kp * N, k = kp * 2;
    float f0 = (k     < Kreal) ? w[(size_t)k * N + n]       : 0.f;
    float f1 = (k + 1 < Kreal) ? w[(size_t)(k + 1) * N + n] : 0.f;
    wp[i] = bfpack(f0, f1);
}

// ft[16,64,64] f32 -> per-graph transposed packed weights [16][32][64]:
// PGW[g][kp][n] = pack(ft[g][n][2kp], ft[g][n][2kp+1])
__global__ void prepack_pg(const float* __restrict__ ft, unsigned* __restrict__ wp, int total) {
    int i = blockIdx.x * blockDim.x + threadIdx.x;
    if (i >= total) return;
    int g = i >> 11, rem = i & 2047, kp = rem >> 6, n = rem & 63;
    const float* f = ft + g * 4096 + n * 64 + kp * 2;
    wp[i] = bfpack(f[0], f[1]);
}

// pos[N,3] -> packed [N,16] uints (K padded 3 -> 32)
__global__ void pack_points3(const float* __restrict__ pos, unsigned* __restrict__ xp, int n) {
    int i = blockIdx.x * blockDim.x + threadIdx.x;
    if (i >= n) return;
    unsigned* o = xp + (size_t)i * 16;
    o[0] = bfpack(pos[3 * i], pos[3 * i + 1]);
    o[1] = bfpack(pos[3 * i + 2], 0.f);
    #pragma unroll
    for (int j = 2; j < 16; ++j) o[j] = 0u;
}

// tp[n] = it[batch[n]] @ pos[n], written packed [N,16]
__global__ void transform3_pack(const float* __restrict__ pos, const int* __restrict__ batch,
                                const float* __restrict__ it9, unsigned* __restrict__ xp, int n) {
    int i = blockIdx.x * blockDim.x + threadIdx.x;
    if (i >= n) return;
    const float* T = it9 + batch[i] * 9;
    float x = pos[3 * i], y = pos[3 * i + 1], z = pos[3 * i + 2];
    float r0 = T[0] * x + T[1] * y + T[2] * z;
    float r1 = T[3] * x + T[4] * y + T[5] * z;
    float r2 = T[6] * x + T[7] * y + T[8] * z;
    unsigned* o = xp + (size_t)i * 16;
    o[0] = bfpack(r0, r1);
    o[1] = bfpack(r2, 0.f);
    #pragma unroll
    for (int j = 2; j < 16; ++j) o[j] = 0u;
}

// Tiny head: out[16,dout] = X[16,K] @ W[K,dout] + b (+identity for 3x3)
__global__ void small_head(const float* __restrict__ X, const float* __restrict__ W,
                           const float* __restrict__ b, float* __restrict__ out0,
                           float* __restrict__ out1, int K, int dout, int addI) {
    int row = blockIdx.x, t = threadIdx.x;
    if (t >= dout) return;
    float s = b[t];
    for (int k = 0; k < K; ++k) s += X[row * K + k] * W[k * dout + t];
    if (addI && (t % 4 == 0)) s += 1.f;       // diag of 3x3: t = 0,4,8
    out0[row * dout + t] = s;
    if (out1) out1[row * dout + t] = s;
}

// ft += I (in place, for PG prepack) and copy to d_out region
__global__ void ft_add_identity(float* __restrict__ ftbuf, float* __restrict__ outp, int total) {
    int i = blockIdx.x * blockDim.x + threadIdx.x;
    if (i >= total) return;
    int loc = i & 4095;
    float v = ftbuf[i] + (((loc >> 6) == (loc & 63)) ? 1.f : 0.f);
    ftbuf[i] = v;
    outp[i]  = v;
}

// ---------------------------------------------------------------------------
// Host side
// ---------------------------------------------------------------------------
namespace {

struct BNL { const float *b, *beta, *gamma, *mean, *var, *w; };

inline BNL bnl(void* const* d, int i) {
    return { (const float*)d[i],     (const float*)d[i + 1], (const float*)d[i + 2],
             (const float*)d[i + 3], (const float*)d[i + 4], (const float*)d[i + 5] };
}

// launch wrappers -----------------------------------------------------------
inline void gBig64(const unsigned* X, const unsigned* W, const BNL& L, unsigned* Y,
                   int M, int K, hipStream_t s) {         // N=64, BN+ReLU, packed out
    gemm_wmma<1, true, true, false, false, false, true>
        <<<dim3(M / 32, 1), dim3(256), 0, s>>>(
            X, W, L.b, L.gamma, L.beta, L.mean, L.var, Y, nullptr, nullptr, M, K, 64);
}
inline void gBig128(const unsigned* X, const unsigned* W, const BNL& L, unsigned* Y,
                    int M, int K, hipStream_t s) {        // N=128, BN+ReLU, packed out
    gemm_wmma<2, true, true, false, false, false, true>
        <<<dim3(M / 32, 1), dim3(256), 0, s>>>(
            X, W, L.b, L.gamma, L.beta, L.mean, L.var, Y, nullptr, nullptr, M, K, 128);
}
inline void gSeg(const unsigned* X, const unsigned* W, const BNL& L, float* G,
                 int M, int K, int N, hipStream_t s) {    // BN+ReLU+segmax, NT=4
    gemm_wmma<4, true, true, true, false, false, false>
        <<<dim3(M / 32, N / 256), dim3(256), 0, s>>>(
            X, W, L.b, L.gamma, L.beta, L.mean, L.var, nullptr, nullptr, (unsigned*)G, M, K, N);
}
inline void gHeadP(const unsigned* X, const unsigned* W, const BNL& L, unsigned* Y,
                   int M, int K, int N, hipStream_t s) {  // M=16, packed out
    gemm_wmma<1, true, true, false, false, true, true>
        <<<dim3((M + 31) / 32, N / 64), dim3(256), 0, s>>>(
            X, W, L.b, L.gamma, L.beta, L.mean, L.var, Y, nullptr, nullptr, M, K, N);
}
inline void gHeadF(const unsigned* X, const unsigned* W, const BNL& L, float* Y,
                   int M, int K, int N, hipStream_t s) {  // M=16, f32 out
    gemm_wmma<1, true, true, false, false, true, false>
        <<<dim3((M + 31) / 32, N / 64), dim3(256), 0, s>>>(
            X, W, L.b, L.gamma, L.beta, L.mean, L.var, nullptr, Y, nullptr, M, K, N);
}

} // namespace

extern "C" void kernel_launch(void* const* d_in, const int* in_sizes, int n_in,
                              void* d_out, int out_size, void* d_ws, size_t ws_size,
                              hipStream_t stream) {
    (void)in_sizes; (void)n_in; (void)out_size; (void)ws_size;
    const int NP = 16 * 4096;
    const int B  = 16;
    hipStream_t s = stream;

    const float* pos   = (const float*)d_in[0];
    const int*   batch = (const int*)  d_in[1];

    // JAX pytree leaf order: sorted dict keys, lists in order,
    // BN layer -> [b, beta, gamma, mean, var, w], plain -> [b, w].
    BNL ftin0  = bnl(d_in, 2),  ftin1  = bnl(d_in, 8),  ftin2 = bnl(d_in, 14);
    BNL ftout0 = bnl(d_in, 20), ftout1 = bnl(d_in, 26);
    const float *ftout2_b = (const float*)d_in[32], *ftout2_w = (const float*)d_in[33];
    BNL itin0  = bnl(d_in, 34), itin1  = bnl(d_in, 40), itin2 = bnl(d_in, 46);
    BNL itout0 = bnl(d_in, 52), itout1 = bnl(d_in, 58);
    const float *itout2_b = (const float*)d_in[64], *itout2_w = (const float*)d_in[65];
    BNL m10 = bnl(d_in, 66), m11 = bnl(d_in, 72);
    BNL m20 = bnl(d_in, 78), m21 = bnl(d_in, 84), m22 = bnl(d_in, 90);
    BNL m30 = bnl(d_in, 96), m31 = bnl(d_in, 102);
    const float *m32_b = (const float*)d_in[108], *m32_w = (const float*)d_in[109];

    // ---- workspace carve-out (uint units) ----
    unsigned* wsu = (unsigned*)d_ws;
    size_t off = 0;
    auto alloc = [&](size_t n) { unsigned* p = wsu + off; off += (n + 15) & ~(size_t)15; return p; };

    unsigned* P0   = alloc((size_t)NP * 16);   // packed K=32 input (pos / transformed)
    unsigned* A64  = alloc((size_t)NP * 32);   // packed [NP,64]
    unsigned* B128 = alloc((size_t)NP * 64);   // packed [NP,128]
    unsigned* X2p  = alloc((size_t)NP * 32);   // x after mlp1
    unsigned* X3p  = alloc((size_t)NP * 32);   // x after feature transform
    float* G1f = (float*)alloc(B * 1024);
    float* G2f = (float*)alloc(B * 1024);
    float* G3f = (float*)alloc(B * 1024);
    unsigned* G1p = alloc(B * 512);
    unsigned* G2p = alloc(B * 512);
    unsigned* G3p = alloc(B * 512);
    unsigned* T1p = alloc(B * 256);
    float*    T2f = (float*)alloc(B * 256);
    unsigned* U1p = alloc(B * 256);
    unsigned* U2p = alloc(B * 128);
    unsigned* V1p = alloc(B * 256);
    float*    V2f = (float*)alloc(B * 256);
    float*    IT9 = (float*)alloc(160);
    float*    FTf = (float*)alloc(B * 4096);
    unsigned* PGW = alloc(B * 2048);           // per-graph transposed packed ft

    // packed weights
    unsigned* Wit0 = alloc(16 * 64);    unsigned* Wit1 = alloc(32 * 128);
    unsigned* Wit2 = alloc(64 * 1024);  unsigned* Wio0 = alloc(512 * 512);
    unsigned* Wio1 = alloc(256 * 256);
    unsigned* Wf0  = alloc(32 * 64);    unsigned* Wf1  = alloc(32 * 128);
    unsigned* Wf2  = alloc(64 * 1024);  unsigned* Wfo0 = alloc(512 * 512);
    unsigned* Wfo1 = alloc(256 * 256);  unsigned* Wfo2 = alloc(128 * 4096);
    unsigned* Wm10 = alloc(16 * 64);    unsigned* Wm11 = alloc(32 * 64);
    unsigned* Wm20 = alloc(32 * 64);    unsigned* Wm21 = alloc(32 * 128);
    unsigned* Wm22 = alloc(64 * 1024);  unsigned* Wm30 = alloc(512 * 512);
    unsigned* Wm31 = alloc(256 * 256);

    float* logits = (float*)d_out;             // [16,5]
    float* itOut  = logits + 80;               // [16,3,3]
    float* ftOut  = itOut + 144;               // [16,64,64]

    auto prep = [&](const float* w, unsigned* wp, int Kr, int Kp, int N) {
        int tot = (Kp >> 1) * N;
        prepack_w<<<dim3((tot + 255) / 256), dim3(256), 0, s>>>(w, wp, Kr, Kp, N);
    };

    // ---- weight prepack (cheap; runs every call; deterministic) ----
    prep(itin0.w, Wit0, 3, 32, 64);      prep(itin1.w, Wit1, 64, 64, 128);
    prep(itin2.w, Wit2, 128, 128, 1024); prep(itout0.w, Wio0, 1024, 1024, 512);
    prep(itout1.w, Wio1, 512, 512, 256);
    prep(ftin0.w, Wf0, 64, 64, 64);      prep(ftin1.w, Wf1, 64, 64, 128);
    prep(ftin2.w, Wf2, 128, 128, 1024);  prep(ftout0.w, Wfo0, 1024, 1024, 512);
    prep(ftout1.w, Wfo1, 512, 512, 256); prep(ftout2_w, Wfo2, 256, 256, 4096);
    prep(m10.w, Wm10, 3, 32, 64);        prep(m11.w, Wm11, 64, 64, 64);
    prep(m20.w, Wm20, 64, 64, 64);       prep(m21.w, Wm21, 64, 64, 128);
    prep(m22.w, Wm22, 128, 128, 1024);   prep(m30.w, Wm30, 1024, 1024, 512);
    prep(m31.w, Wm31, 512, 512, 256);

    // ---- input T-Net ----
    pack_points3<<<dim3(NP / 256), dim3(256), 0, s>>>(pos, P0, NP);
    gBig64(P0, Wit0, itin0, A64, NP, 32, s);
    gBig128(A64, Wit1, itin1, B128, NP, 64, s);
    fill_zero<<<dim3(64), dim3(256), 0, s>>>(G1f, B * 1024);
    gSeg(B128, Wit2, itin2, G1f, NP, 128, 1024, s);
    pack_rows<<<dim3(32), dim3(256), 0, s>>>(G1f, G1p, B * 512);
    gHeadP(G1p, Wio0, itout0, T1p, B, 1024, 512, s);
    gHeadF(T1p, Wio1, itout1, T2f, B, 512, 256, s);
    small_head<<<dim3(B), dim3(32), 0, s>>>(T2f, itout2_w, itout2_b, IT9, itOut, 256, 9, 1);
    transform3_pack<<<dim3(NP / 256), dim3(256), 0, s>>>(pos, batch, IT9, P0, NP);

    // ---- mlp1 ----
    gBig64(P0, Wm10, m10, A64, NP, 32, s);
    gBig64(A64, Wm11, m11, X2p, NP, 64, s);

    // ---- feature T-Net ----
    gBig64(X2p, Wf0, ftin0, A64, NP, 64, s);
    gBig128(A64, Wf1, ftin1, B128, NP, 64, s);
    fill_zero<<<dim3(64), dim3(256), 0, s>>>(G2f, B * 1024);
    gSeg(B128, Wf2, ftin2, G2f, NP, 128, 1024, s);
    pack_rows<<<dim3(32), dim3(256), 0, s>>>(G2f, G2p, B * 512);
    gHeadP(G2p, Wfo0, ftout0, U1p, B, 1024, 512, s);
    gHeadP(U1p, Wfo1, ftout1, U2p, B, 512, 256, s);
    // plain 256 -> 4096, f32 out
    gemm_wmma<1, false, false, false, false, true, false>
        <<<dim3(1, 4096 / 64), dim3(256), 0, s>>>(
            U2p, Wfo2, ftout2_b, nullptr, nullptr, nullptr, nullptr,
            nullptr, FTf, nullptr, B, 256, 4096);
    ft_add_identity<<<dim3(B * 4096 / 256), dim3(256), 0, s>>>(FTf, ftOut, B * 4096);
    prepack_pg<<<dim3(B * 2048 / 256), dim3(256), 0, s>>>(FTf, PGW, B * 2048);

    // per-graph 64x64 feature transform (WMMA, PG mode, packed out)
    gemm_wmma<1, false, false, false, true, false, true>
        <<<dim3(NP / 32, 1), dim3(256), 0, s>>>(
            X2p, PGW, nullptr, nullptr, nullptr, nullptr, nullptr,
            X3p, nullptr, nullptr, NP, 64, 64);

    // ---- mlp2 + global max ----
    gBig64(X3p, Wm20, m20, A64, NP, 64, s);
    gBig128(A64, Wm21, m21, B128, NP, 64, s);
    fill_zero<<<dim3(64), dim3(256), 0, s>>>(G3f, B * 1024);
    gSeg(B128, Wm22, m22, G3f, NP, 128, 1024, s);
    pack_rows<<<dim3(32), dim3(256), 0, s>>>(G3f, G3p, B * 512);

    // ---- mlp3 head ----
    gHeadP(G3p, Wm30, m30, V1p, B, 1024, 512, s);
    gHeadF(V1p, Wm31, m31, V2f, B, 512, 256, s);
    small_head<<<dim3(B), dim3(32), 0, s>>>(V2f, m32_w, m32_b, logits, nullptr, 256, 5, 0);
}
